// ToyTransformerBlock_76733885710751
// MI455X (gfx1250) — matrix operand order repaired and checked by reading, not compiled, not run on hardware
//
#include <hip/hip_runtime.h>
#include <hip/hip_bf16.h>
#include <math.h>

typedef unsigned short ushort_t;
typedef __attribute__((ext_vector_type(16))) __bf16 v16bf;
typedef __attribute__((ext_vector_type(8)))  __bf16 v8bf;
typedef __attribute__((ext_vector_type(8)))  float  v8f;

#define D_MODEL  2048
#define N_HEADS  16
#define HEAD_DIM 128
#define D_FF     8192
#define SEQ      2048
#define BATCH    2
#define ROWS     (BATCH * SEQ)   /* 4096 */

// ---------- small helpers ----------
static __device__ __forceinline__ ushort_t f2bf(float f) {
  unsigned u = __builtin_bit_cast(unsigned, f);
  u += 0x7fffu + ((u >> 16) & 1u);          // round-to-nearest-even
  return (ushort_t)(u >> 16);
}
static __device__ __forceinline__ v16bf cat8(v8bf lo, v8bf hi) {
  v16bf r;
#pragma unroll
  for (int i = 0; i < 8; ++i) { r[i] = lo[i]; r[i + 8] = hi[i]; }
  return r;
}
// A-fragment 16x32 bf16 (row-major src, ld elems). p = tile origin (m0,k0).
// Lane L: row = L&15; halves at k = (L>>4)*8 + [0..7] and 16 + (L>>4)*8 + [0..7].
static __device__ __forceinline__ v16bf load_a_frag(const ushort_t* p, int ld, int lane) {
  const int r = lane & 15, h = lane >> 4;
  const ushort_t* q = p + (size_t)r * ld + h * 8;
  v8bf lo = *reinterpret_cast<const v8bf*>(q);
  v8bf hi = *reinterpret_cast<const v8bf*>(q + 16);
  return cat8(lo, hi);
}
// B-fragment 32x16 sourced from row-major W[N,K]. p = tile origin (n0,k0).
// Lane L: column n = L&15 (a row of W); halves at k = (L>>4)*8 + [0..7] and 16 + (L>>4)*8 + [0..7].
static __device__ __forceinline__ v16bf load_b_frag(const ushort_t* p, int ld, int lane) {
  const int r = lane & 15, h = lane >> 4;
  const ushort_t* q = p + (size_t)r * ld + h * 8;
  v8bf lo = *reinterpret_cast<const v8bf*>(q);
  v8bf hi = *reinterpret_cast<const v8bf*>(q + 16);
  return cat8(lo, hi);
}
static __device__ __forceinline__ v8f wmma_bf16(v16bf a, v16bf b, v8f c) {
  return __builtin_amdgcn_wmma_f32_16x16x32_bf16(false, a, false, b, (short)0, c, false, false);
}

// ---------- elementwise kernels ----------
__global__ void cast_f32_bf16(const float* __restrict__ in, ushort_t* __restrict__ out, int n) {
  int i = blockIdx.x * 256 + threadIdx.x;
  if (i < n) out[i] = f2bf(in[i]);
}

__global__ void transpose_v(const ushort_t* __restrict__ v, ushort_t* __restrict__ vt) {
  int i = blockIdx.x * 256 + threadIdx.x;           // over ROWS*D_MODEL
  if (i >= ROWS * D_MODEL) return;
  int col = i % D_MODEL, row = i / D_MODEL;         // row = b*SEQ + t
  int b = row / SEQ, t = row % SEQ;
  int h = col / HEAD_DIM, d = col % HEAD_DIM;
  vt[((size_t)(b * N_HEADS + h) * HEAD_DIM + d) * SEQ + t] = v[i];
}

__global__ __launch_bounds__(256) void layernorm_bf16(
    const float* __restrict__ x, const float* __restrict__ g, const float* __restrict__ b,
    ushort_t* __restrict__ out, int C) {
  __shared__ float red[256];
  const int row = blockIdx.x, tid = threadIdx.x;
  const float* xr = x + (size_t)row * C;
  float s = 0.f;
  for (int i = tid; i < C; i += 256) s += xr[i];
  red[tid] = s; __syncthreads();
  for (int off = 128; off > 0; off >>= 1) { if (tid < off) red[tid] += red[tid + off]; __syncthreads(); }
  const float mu = red[0] / C;
  __syncthreads();
  float v = 0.f;
  for (int i = tid; i < C; i += 256) { float d = xr[i] - mu; v += d * d; }
  red[tid] = v; __syncthreads();
  for (int off = 128; off > 0; off >>= 1) { if (tid < off) red[tid] += red[tid + off]; __syncthreads(); }
  const float rstd = rsqrtf(red[0] / C + 1e-5f);
  ushort_t* o = out + (size_t)row * C;
  for (int i = tid; i < C; i += 256) o[i] = f2bf((xr[i] - mu) * rstd * g[i] + b[i]);
}

// ---------- WMMA GEMM: C[M,N] = A[M,K](bf16) @ W[N,K]^T(bf16) ----------
// Block tile 128x256 (8 waves as 2x4), wave tile 64x64 (16 WMMA tiles).
// EPI: 0 = f32 store
//      1 = bf16 store
//      2 = f32 store + residual add (aux = residual)
//      3 = bf16 store of silu(aux)*acc  (SwiGLU fusion; aux = gate f32)
template <int EPI>
__global__ __launch_bounds__(256) void gemm_bf16_wmma(
    const ushort_t* __restrict__ A, const ushort_t* __restrict__ W,
    float* __restrict__ Cf, ushort_t* __restrict__ Cb,
    const float* __restrict__ aux, int N, int K) {
  const int lane = threadIdx.x & 31;
  const int wave = threadIdx.x >> 5;
  const int m0 = blockIdx.x * 128 + (wave >> 2) * 64;   // 2 wave rows * 64
  const int n0 = blockIdx.y * 256 + (wave & 3) * 64;    // 4 wave cols * 64
  const ushort_t* Ap = A + (size_t)m0 * K;
  const ushort_t* Wp = W + (size_t)n0 * K;

  v8f acc[4][4];
#pragma unroll
  for (int i = 0; i < 4; ++i)
#pragma unroll
    for (int j = 0; j < 4; ++j)
#pragma unroll
      for (int e = 0; e < 8; ++e) acc[i][j][e] = 0.f;

  for (int k0 = 0; k0 < K; k0 += 32) {
    v16bf af[4];
#pragma unroll
    for (int i = 0; i < 4; ++i)
      af[i] = load_a_frag(Ap + (size_t)(i * 16) * K + k0, K, lane);
    // one B fragment live at a time keeps VGPR pressure ~190
#pragma unroll
    for (int j = 0; j < 4; ++j) {
      v16bf wf = load_b_frag(Wp + (size_t)(j * 16) * K + k0, K, lane);
#pragma unroll
      for (int i = 0; i < 4; ++i)
        acc[i][j] = wmma_bf16(af[i], wf, acc[i][j]);
    }
    // prefetch next-next k tile (global_prefetch_b8); 32 lanes cover 64 rows x2
    if (k0 + 64 < K) {
      __builtin_prefetch(Ap + (size_t)lane * K + k0 + 64, 0, 3);
      __builtin_prefetch(Ap + (size_t)(32 + lane) * K + k0 + 64, 0, 3);
      __builtin_prefetch(Wp + (size_t)lane * K + k0 + 64, 0, 3);
      __builtin_prefetch(Wp + (size_t)(32 + lane) * K + k0 + 64, 0, 3);
    }
  }

  const int cn = lane & 15, ch = lane >> 4;
#pragma unroll
  for (int i = 0; i < 4; ++i)
#pragma unroll
    for (int j = 0; j < 4; ++j)
#pragma unroll
      for (int e = 0; e < 8; ++e) {
        const int row = m0 + i * 16 + ch * 8 + e;
        const int col = n0 + j * 16 + cn;
        const size_t idx = (size_t)row * N + col;
        float v = acc[i][j][e];
        if constexpr (EPI == 0) {
          Cf[idx] = v;
        } else if constexpr (EPI == 1) {
          Cb[idx] = f2bf(v);
        } else if constexpr (EPI == 2) {
          Cf[idx] = v + aux[idx];
        } else {  // EPI == 3: silu(gate) * up -> bf16
          float g = aux[idx];
          float sg = g / (1.f + __expf(-g));
          Cb[idx] = f2bf(sg * v);
        }
      }
}

// ---------- flash attention: 1 wave per (b,h,16-row q tile) ----------
__global__ __launch_bounds__(32) void attn_wmma(
    const ushort_t* __restrict__ Q, const ushort_t* __restrict__ Kc,
    const ushort_t* __restrict__ Vt, ushort_t* __restrict__ O) {
  __shared__ ushort_t pshm[16][32];
  const int lane = threadIdx.x;
  const int q0 = blockIdx.x * 16;
  const int bh = blockIdx.y;
  const int b = bh >> 4, h = bh & 15;
  const ushort_t* qbase = Q  + ((size_t)(b * SEQ + q0)) * D_MODEL + h * HEAD_DIM;
  const ushort_t* kbase = Kc + ((size_t)(b * SEQ)) * D_MODEL + h * HEAD_DIM;
  const ushort_t* vbase = Vt + (size_t)bh * HEAD_DIM * SEQ;

  v16bf qf[4];
#pragma unroll
  for (int j = 0; j < 4; ++j) qf[j] = load_a_frag(qbase + j * 32, D_MODEL, lane);

  v8f o[8];
#pragma unroll
  for (int n = 0; n < 8; ++n)
#pragma unroll
    for (int e = 0; e < 8; ++e) o[n][e] = 0.f;

  float mrun[8], lrun[8];
#pragma unroll
  for (int e = 0; e < 8; ++e) { mrun[e] = -1e30f; lrun[e] = 0.f; }

  const float scale = 0.08838834764831845f;   // 1/sqrt(128)
  const int r = lane & 15, hh = lane >> 4;

  for (int t0 = 0; t0 < SEQ; t0 += 32) {
    v8f s0, s1;
#pragma unroll
    for (int e = 0; e < 8; ++e) { s0[e] = 0.f; s1[e] = 0.f; }
#pragma unroll
    for (int j = 0; j < 4; ++j) {
      v16bf kf = load_b_frag(kbase + (size_t)t0 * D_MODEL + j * 32, D_MODEL, lane);
      s0 = wmma_bf16(qf[j], kf, s0);
    }
#pragma unroll
    for (int j = 0; j < 4; ++j) {
      v16bf kf = load_b_frag(kbase + (size_t)(t0 + 16) * D_MODEL + j * 32, D_MODEL, lane);
      s1 = wmma_bf16(qf[j], kf, s1);
    }
    // online softmax: element e lives at (row = hh*8+e, col = r / 16+r);
    // each row's 16 columns span exactly one 16-lane group -> width-16 shfl reduce.
    float fac[8];
#pragma unroll
    for (int e = 0; e < 8; ++e) {
      float a0 = s0[e] * scale, a1 = s1[e] * scale;
      float mx = fmaxf(a0, a1);
      mx = fmaxf(mx, __shfl_xor(mx, 1, 16));
      mx = fmaxf(mx, __shfl_xor(mx, 2, 16));
      mx = fmaxf(mx, __shfl_xor(mx, 4, 16));
      mx = fmaxf(mx, __shfl_xor(mx, 8, 16));
      float mnew = fmaxf(mrun[e], mx);
      fac[e] = __expf(mrun[e] - mnew);
      float p0 = __expf(a0 - mnew), p1 = __expf(a1 - mnew);
      float ls = p0 + p1;
      ls += __shfl_xor(ls, 1, 16);
      ls += __shfl_xor(ls, 2, 16);
      ls += __shfl_xor(ls, 4, 16);
      ls += __shfl_xor(ls, 8, 16);
      lrun[e] = lrun[e] * fac[e] + ls;
      mrun[e] = mnew;
      s0[e] = p0; s1[e] = p1;
    }
#pragma unroll
    for (int n = 0; n < 8; ++n)
#pragma unroll
      for (int e = 0; e < 8; ++e) o[n][e] *= fac[e];

    // reshape P: C-layout -> A-layout via LDS (bf16)
    __syncthreads();
#pragma unroll
    for (int e = 0; e < 8; ++e) {
      pshm[hh * 8 + e][r]      = f2bf(s0[e]);
      pshm[hh * 8 + e][16 + r] = f2bf(s1[e]);
    }
    __syncthreads();
    // A-layout fragment = two contiguous 16B runs of row r -> 2x ds_load_b128
    v8bf plo = *reinterpret_cast<const v8bf*>(&pshm[r][hh * 8]);
    v8bf phi = *reinterpret_cast<const v8bf*>(&pshm[r][16 + hh * 8]);
    v16bf pf = cat8(plo, phi);
    __syncthreads();
#pragma unroll
    for (int n = 0; n < 8; ++n) {
      v16bf vf = load_b_frag(vbase + (size_t)(n * 16) * SEQ + t0, SEQ, lane);
      o[n] = wmma_bf16(pf, vf, o[n]);
    }
  }

  float linv[8];
#pragma unroll
  for (int e = 0; e < 8; ++e) linv[e] = 1.f / lrun[e];
#pragma unroll
  for (int n = 0; n < 8; ++n)
#pragma unroll
    for (int e = 0; e < 8; ++e) {
      float v = o[n][e] * linv[e];
      const int row = q0 + hh * 8 + e;
      const int col = h * HEAD_DIM + n * 16 + r;
      O[((size_t)(b * SEQ + row)) * D_MODEL + col] = f2bf(v);
    }
}

// ---------- orchestration ----------
extern "C" void kernel_launch(void* const* d_in, const int* in_sizes, int n_in,
                              void* d_out, int out_size, void* d_ws, size_t ws_size,
                              hipStream_t stream) {
  const float* x    = (const float*)d_in[0];
  const float* wq   = (const float*)d_in[1];
  const float* wk   = (const float*)d_in[2];
  const float* wv   = (const float*)d_in[3];
  const float* wo   = (const float*)d_in[4];
  const float* wg   = (const float*)d_in[5];
  const float* wu   = (const float*)d_in[6];
  const float* wd   = (const float*)d_in[7];
  const float* ln1w = (const float*)d_in[8];
  const float* ln1b = (const float*)d_in[9];
  const float* ln2w = (const float*)d_in[10];
  const float* ln2b = (const float*)d_in[11];

  char* ws = (char*)d_ws;
  size_t off = 0;
  auto alloc = [&](size_t bytes) -> void* {
    void* p = ws + off;
    off += (bytes + 255) & ~(size_t)255;
    return p;
  };
  const size_t SQ  = (size_t)D_MODEL * D_MODEL;   // square weight elems
  const size_t FF  = (size_t)D_FF * D_MODEL;      // ffn weight elems
  const size_t ACT = (size_t)ROWS * D_MODEL;      // activation elems
  const size_t FFA = (size_t)ROWS * D_FF;         // ffn activation elems

  ushort_t* wqb  = (ushort_t*)alloc(SQ * 2);
  ushort_t* wkb  = (ushort_t*)alloc(SQ * 2);
  ushort_t* wvb  = (ushort_t*)alloc(SQ * 2);
  ushort_t* wob  = (ushort_t*)alloc(SQ * 2);
  ushort_t* wgb  = (ushort_t*)alloc(FF * 2);
  ushort_t* wub  = (ushort_t*)alloc(FF * 2);
  ushort_t* wdb  = (ushort_t*)alloc(FF * 2);
  ushort_t* ln1o = (ushort_t*)alloc(ACT * 2);
  ushort_t* qb   = (ushort_t*)alloc(ACT * 2);
  ushort_t* kb   = (ushort_t*)alloc(ACT * 2);
  ushort_t* vb   = (ushort_t*)alloc(ACT * 2);
  ushort_t* vtb  = (ushort_t*)alloc(ACT * 2);
  ushort_t* ab   = (ushort_t*)alloc(ACT * 2);
  float*    x1   = (float*)   alloc(ACT * 4);
  ushort_t* ln2o = (ushort_t*)alloc(ACT * 2);
  float*    gf   = (float*)   alloc(FFA * 4);
  ushort_t* hb   = (ushort_t*)alloc(FFA * 2);

  // 1) weights -> bf16
  cast_f32_bf16<<<(int)((SQ + 255) / 256), 256, 0, stream>>>(wq, wqb, (int)SQ);
  cast_f32_bf16<<<(int)((SQ + 255) / 256), 256, 0, stream>>>(wk, wkb, (int)SQ);
  cast_f32_bf16<<<(int)((SQ + 255) / 256), 256, 0, stream>>>(wv, wvb, (int)SQ);
  cast_f32_bf16<<<(int)((SQ + 255) / 256), 256, 0, stream>>>(wo, wob, (int)SQ);
  cast_f32_bf16<<<(int)((FF + 255) / 256), 256, 0, stream>>>(wg, wgb, (int)FF);
  cast_f32_bf16<<<(int)((FF + 255) / 256), 256, 0, stream>>>(wu, wub, (int)FF);
  cast_f32_bf16<<<(int)((FF + 255) / 256), 256, 0, stream>>>(wd, wdb, (int)FF);

  // 2) LN1
  layernorm_bf16<<<ROWS, 256, 0, stream>>>(x, ln1w, ln1b, ln1o, D_MODEL);

  // 3) Q,K,V projections (bf16 out)
  dim3 gSq(ROWS / 128, D_MODEL / 256);
  gemm_bf16_wmma<1><<<gSq, 256, 0, stream>>>(ln1o, wqb, nullptr, qb, nullptr, D_MODEL, D_MODEL);
  gemm_bf16_wmma<1><<<gSq, 256, 0, stream>>>(ln1o, wkb, nullptr, kb, nullptr, D_MODEL, D_MODEL);
  gemm_bf16_wmma<1><<<gSq, 256, 0, stream>>>(ln1o, wvb, nullptr, vb, nullptr, D_MODEL, D_MODEL);

  // 4) V -> Vt[b,h,d,t]
  transpose_v<<<(int)((ACT + 255) / 256), 256, 0, stream>>>(vb, vtb);

  // 5) flash attention
  dim3 gAt(SEQ / 16, BATCH * N_HEADS);
  attn_wmma<<<gAt, 32, 0, stream>>>(qb, kb, vtb, ab);

  // 6) output projection + residual -> x1 (f32)
  gemm_bf16_wmma<2><<<gSq, 256, 0, stream>>>(ab, wob, x1, nullptr, x, D_MODEL, D_MODEL);

  // 7) LN2
  layernorm_bf16<<<ROWS, 256, 0, stream>>>(x1, ln2w, ln2b, ln2o, D_MODEL);

  // 8) gate projection (f32), then up projection with fused SwiGLU -> bf16 h
  dim3 gFf(ROWS / 128, D_FF / 256);
  gemm_bf16_wmma<0><<<gFf, 256, 0, stream>>>(ln2o, wgb, gf, nullptr, nullptr, D_FF, D_MODEL);
  gemm_bf16_wmma<3><<<gFf, 256, 0, stream>>>(ln2o, wub, nullptr, hb, gf, D_FF, D_MODEL);

  // 9) down projection + residual -> d_out (f32)
  gemm_bf16_wmma<2><<<gSq, 256, 0, stream>>>(hb, wdb, (float*)d_out, nullptr, x1, D_MODEL, D_FF);
}
